// MoE_HyperModel_51823075393998
// MI455X (gfx1250) — matrix operand order repaired and checked by reading, not compiled, not run on hardware
//
#include <hip/hip_runtime.h>
#include <hip/hip_bf16.h>

// ---------------- types ----------------
typedef __attribute__((ext_vector_type(16))) __bf16 v16bf;
typedef __attribute__((ext_vector_type(8)))  __bf16 v8bf;
typedef __attribute__((ext_vector_type(8)))  float  v8f;
typedef __attribute__((ext_vector_type(4)))  int    v4i;

__device__ __forceinline__ __bf16 f2bf(float f) {
    union { unsigned u; float f; } v; v.f = f;
    unsigned r = v.u + 0x7FFFu + ((v.u >> 16) & 1u);
    union { unsigned short s; __bf16 b; } o; o.s = (unsigned short)(r >> 16);
    return o.b;
}

// ---------------- CDNA5 async global->LDS (probe via __has_builtin) ----------------
#if defined(__has_builtin)
#if __has_builtin(__builtin_amdgcn_global_load_async_to_lds_b128)
#define HAVE_ASYNC_B128 1
#else
#define HAVE_ASYNC_B128 0
#endif
#else
#define HAVE_ASYNC_B128 0
#endif

#if HAVE_ASYNC_B128
typedef __attribute__((address_space(1))) v4i gv4i;   // global 128-bit chunk
typedef __attribute__((address_space(3))) v4i lv4i;   // LDS 128-bit chunk
#endif

__device__ __forceinline__ void async_copy16(const __bf16* g, __bf16* l) {
#if HAVE_ASYNC_B128
    __builtin_amdgcn_global_load_async_to_lds_b128(
        (gv4i*)(void*)g, (lv4i*)(void*)l, 0, 0);
#else
    (void)g; (void)l;
#endif
}

__device__ __forceinline__ void wait_async0() {
#if defined(__has_builtin) && __has_builtin(__builtin_amdgcn_s_wait_asynccnt)
    __builtin_amdgcn_s_wait_asynccnt(0);
#else
    asm volatile("s_wait_asynccnt 0x0" ::: "memory");
#endif
}

// ---------------- problem dims ----------------
static constexpr int Bb   = 4096;   // batch
static constexpr int Dd   = 1024;   // embedding
static constexpr int Cc   = 512;    // context
static constexpr int Ee   = 16;     // experts
static constexpr int DC   = Dd + Cc;        // 1536
static constexpr int HG   = 2048;   // gating hidden
static constexpr int HE   = 2048;   // expert hidden
static constexpr int Zz   = 512;    // expert out
static constexpr int NH   = 16384;  // LIN*R == R*LOUT

// ---------------- elementwise / transpose kernels ----------------
__global__ void cvt_f32_bf16(const float* __restrict__ in, __bf16* __restrict__ out, long long n) {
    long long i = (long long)blockIdx.x * blockDim.x + threadIdx.x;
    if (i < n) out[i] = f2bf(in[i]);
}

// in: (z, K, N) f32 row-major -> out: (z, N, K) bf16 (transposed per z-slice)
__global__ void cvt_transpose_bf16(const float* __restrict__ in, __bf16* __restrict__ out,
                                   int K, int N) {
    __shared__ float tile[32][33];
    in  += (size_t)blockIdx.z * K * N;
    out += (size_t)blockIdx.z * K * N;
    int k0 = blockIdx.y * 32, n0 = blockIdx.x * 32;
    #pragma unroll
    for (int r = threadIdx.y; r < 32; r += 8) {
        int k = k0 + r, n = n0 + threadIdx.x;
        tile[r][threadIdx.x] = (k < K && n < N) ? in[(long long)k * N + n] : 0.0f;
    }
    __syncthreads();
    #pragma unroll
    for (int r = threadIdx.y; r < 32; r += 8) {
        int n = n0 + r, k = k0 + threadIdx.x;
        if (n < N && k < K) out[(long long)n * K + k] = f2bf(tile[threadIdx.x][r]);
    }
}

__global__ void concat_bf16(const float* __restrict__ s, const float* __restrict__ c,
                            __bf16* __restrict__ x, long long n) {
    long long i = (long long)blockIdx.x * blockDim.x + threadIdx.x;
    if (i >= n) return;
    int row = (int)(i / DC), col = (int)(i % DC);
    float v = (col < Dd) ? s[(long long)row * Dd + col] : c[(long long)row * Cc + (col - Dd)];
    x[i] = f2bf(v);
}

__global__ void zero_f32(float* __restrict__ p, long long n) {
    long long i = (long long)blockIdx.x * blockDim.x + threadIdx.x;
    if (i < n) p[i] = 0.0f;
}

__global__ void softmax16(const float* __restrict__ logits, float* __restrict__ gates, int rows) {
    int b = blockIdx.x * blockDim.x + threadIdx.x;
    if (b >= rows) return;
    float v[16], mx = -3.4e38f;
    #pragma unroll
    for (int i = 0; i < 16; i++) { v[i] = logits[b * 16 + i]; mx = fmaxf(mx, v[i]); }
    float sum = 0.0f;
    #pragma unroll
    for (int i = 0; i < 16; i++) { v[i] = __expf(v[i] - mx); sum += v[i]; }
    float inv = 1.0f / sum;
    #pragma unroll
    for (int i = 0; i < 16; i++) gates[b * 16 + i] = v[i] * inv;
}

// ---------------- WMMA GEMM (double-buffered pipeline) ----------------
// out(MxN) = A(MxK bf16, ld=lda) @ Wt^T  where Wt is N x K bf16 (pre-transposed)
// MODE 0: outH = bf16(relu(acc + bias[n]))
// MODE 1: outF[m,n] += gate[m*gs] * (acc + bias[n])   (fused soft-MoE reduce)
// MODE 2: outF[m,n]  = acc + bias[n]
// Block tile 128x128, BK=32, 8 wave32 waves (2 M x 4 N), wave tile 64x32.
template <int MODE, bool NGUARD>
__global__ __launch_bounds__(256, 2)
void gemm_bf16_wmma(const __bf16* __restrict__ A, int lda,
                    const __bf16* __restrict__ Wt,          // N x K row-major
                    const float* __restrict__ bias,
                    float* __restrict__ outF, __bf16* __restrict__ outH, int ldo,
                    const float* __restrict__ gate, int gs,
                    int M, int N, int K) {
    __shared__ alignas(16) __bf16 sA[2][128 * 32];   // [m][k]
    __shared__ alignas(16) __bf16 sB[2][128 * 32];   // [n][k]

    const int tid  = threadIdx.x;
    const int lane = tid & 31;
    const int wid  = tid >> 5;
    const int waveM = wid & 1;
    const int waveN = wid >> 1;
    const int mBlock = blockIdx.y * 128;
    const int nBlock = blockIdx.x * 128;

    // fill coordinates: each thread copies 2x16B for A and 2x16B for B per tile
    const int fRow = tid >> 2;           // 0..63
    const int fK   = (tid & 3) << 3;     // 0,8,16,24

    constexpr bool USE_ASYNC = (HAVE_ASYNC_B128 != 0) && !NGUARD;

    v8f acc[4][2];
    #pragma unroll
    for (int i = 0; i < 4; i++)
        #pragma unroll
        for (int j = 0; j < 2; j++)
            acc[i][j] = v8f{0.f, 0.f, 0.f, 0.f, 0.f, 0.f, 0.f, 0.f};

    const int aRow = waveM * 64 + (lane & 15);
    const int aKb  = (lane < 16) ? 0 : 8;    // 16-bit A 16x32 lane layout
    const int bCol = waveN * 32 + (lane & 15);
    const int bKb  = (lane < 16) ? 0 : 8;    // 16-bit B 32x16 lane layout

    v8bf ra[2], rb[2];
    auto gA = [&](int t, int h) { return A  + (size_t)(mBlock + fRow + 64 * h) * lda + t * 32 + fK; };
    auto gB = [&](int t, int h) { return Wt + (size_t)(nBlock + fRow + 64 * h) * K   + t * 32 + fK; };

    auto loadRegs = [&](int t) {
        #pragma unroll
        for (int h = 0; h < 2; h++) {
            ra[h] = *(const v8bf*)gA(t, h);
            if (NGUARD && (nBlock + fRow + 64 * h) >= N) {
                #pragma unroll
                for (int q = 0; q < 8; q++) rb[h][q] = f2bf(0.0f);
            } else {
                rb[h] = *(const v8bf*)gB(t, h);
            }
            if ((t + 1) * 32 < K) {
                __builtin_prefetch(gA(t, h) + 32, 0, 0);      // global_prefetch_b8
                if (!NGUARD) __builtin_prefetch(gB(t, h) + 32, 0, 0);
            }
        }
    };
    auto storeRegs = [&](int buf) {
        #pragma unroll
        for (int h = 0; h < 2; h++) {
            *(v8bf*)&sA[buf][(fRow + 64 * h) * 32 + fK] = ra[h];
            *(v8bf*)&sB[buf][(fRow + 64 * h) * 32 + fK] = rb[h];
        }
    };
    auto asyncFill = [&](int t, int buf) {
        #pragma unroll
        for (int h = 0; h < 2; h++) {
            async_copy16(gA(t, h), &sA[buf][(fRow + 64 * h) * 32 + fK]);
            async_copy16(gB(t, h), &sB[buf][(fRow + 64 * h) * 32 + fK]);
        }
    };

    const int T = K >> 5;
    // prologue: stage tile 0 into buffer 0
    if constexpr (USE_ASYNC) {
        asyncFill(0, 0);
        wait_async0();
    } else {
        loadRegs(0);
        storeRegs(0);
    }
    __syncthreads();

    for (int t = 0; t < T; ++t) {
        const int cur = t & 1;
        const bool more = (t + 1 < T);

        // issue next tile's fill (overlaps with WMMA compute below)
        if (more) {
            if constexpr (USE_ASYNC) asyncFill(t + 1, cur ^ 1);
            else                     loadRegs(t + 1);
        }

        // ---- fragments + 8 WMMAs from current buffer ----
        v16bf af[4];
        #pragma unroll
        for (int i = 0; i < 4; i++) {
            const __bf16* p = &sA[cur][(aRow + i * 16) * 32 + aKb];
            v8bf lo = *(const v8bf*)p;
            v8bf hi = *(const v8bf*)(p + 16);
            af[i] = __builtin_shufflevector(lo, hi, 0,1,2,3,4,5,6,7,8,9,10,11,12,13,14,15);
        }
        v16bf bfr[2];
        #pragma unroll
        for (int j = 0; j < 2; j++) {
            const __bf16* p = &sB[cur][(bCol + j * 16) * 32 + bKb];
            v8bf lo = *(const v8bf*)p;
            v8bf hi = *(const v8bf*)(p + 16);
            bfr[j] = __builtin_shufflevector(lo, hi, 0,1,2,3,4,5,6,7,8,9,10,11,12,13,14,15);
        }
        #pragma unroll
        for (int i = 0; i < 4; i++)
            #pragma unroll
            for (int j = 0; j < 2; j++)
                acc[i][j] = __builtin_amdgcn_wmma_f32_16x16x32_bf16(
                    false, af[i], false, bfr[j], (short)0, acc[i][j], false, false);

        // commit next buffer
        if (more) {
            if constexpr (USE_ASYNC) wait_async0();
            else                     storeRegs(cur ^ 1);
            __syncthreads();
        }
    }

    // ---- epilogue: C frag: col = lane&15, row = r + 8*(lane>=16) ----
    const int laneHi8 = (lane >> 4) << 3;
    #pragma unroll
    for (int i = 0; i < 4; i++) {
        #pragma unroll
        for (int j = 0; j < 2; j++) {
            #pragma unroll
            for (int r = 0; r < 8; r++) {
                int gr = mBlock + waveM * 64 + i * 16 + r + laneHi8;
                int gc = nBlock + waveN * 32 + j * 16 + (lane & 15);
                if (NGUARD && gc >= N) continue;
                float v = acc[i][j][r] + bias[gc];
                if (MODE == 0) {
                    outH[(long long)gr * ldo + gc] = f2bf(fmaxf(v, 0.0f));
                } else if (MODE == 1) {
                    outF[(long long)gr * ldo + gc] += gate[(long long)gr * gs] * v;
                } else {
                    outF[(long long)gr * ldo + gc] = v;
                }
            }
        }
    }
}

// ---------------- host orchestration ----------------
static inline long long cdiv(long long a, long long b) { return (a + b - 1) / b; }

extern "C" void kernel_launch(void* const* d_in, const int* in_sizes, int n_in,
                              void* d_out, int out_size, void* d_ws, size_t ws_size,
                              hipStream_t stream) {
    const float* s   = (const float*)d_in[0];
    const float* c   = (const float*)d_in[1];
    const float* Wg1 = (const float*)d_in[2];
    const float* bg1 = (const float*)d_in[3];
    const float* Wg2 = (const float*)d_in[4];
    const float* bg2 = (const float*)d_in[5];
    const float* We1 = (const float*)d_in[6];
    const float* be1 = (const float*)d_in[7];
    const float* We2 = (const float*)d_in[8];
    const float* be2 = (const float*)d_in[9];
    const float* WA  = (const float*)d_in[10];
    const float* bA  = (const float*)d_in[11];
    const float* WB  = (const float*)d_in[12];
    const float* bB  = (const float*)d_in[13];
    float* out = (float*)d_out;

    // workspace carve-up (256B aligned). Weight buffers hold TRANSPOSED (N x K) bf16.
    char* ws = (char*)d_ws;
    size_t off = 0;
    auto carve = [&](size_t bytes) { char* p = ws + off; off += (bytes + 255) & ~(size_t)255; return p; };
    __bf16* xb    = (__bf16*)carve((size_t)Bb * DC * 2);
    __bf16* g1    = (__bf16*)carve((size_t)Bb * HG * 2);
    __bf16* h     = (__bf16*)carve((size_t)Bb * HE * 2);
    __bf16* Wg1t  = (__bf16*)carve((size_t)Dd * HG * 2);
    __bf16* Wg2t  = (__bf16*)carve((size_t)HG * Ee * 2);
    __bf16* We1t  = (__bf16*)carve((size_t)Ee * DC * HE * 2);
    __bf16* We2t  = (__bf16*)carve((size_t)Ee * HE * Zz * 2);
    __bf16* WAt   = (__bf16*)carve((size_t)Zz * NH * 2);
    __bf16* WBt   = (__bf16*)carve((size_t)Zz * NH * 2);
    float*  logit = (float*)carve((size_t)Bb * Ee * 4);
    float*  gates = (float*)carve((size_t)Bb * Ee * 4);
    float*  z     = (float*)carve((size_t)Bb * Zz * 4);
    __bf16* zb    = (__bf16*)carve((size_t)Bb * Zz * 2);
    (void)ws_size; (void)in_sizes; (void)n_in; (void)out_size;

    const int T = 256;
    auto TRT = [&](const float* src, __bf16* dst, int K, int N, int E_) {
        dim3 g((unsigned)cdiv(N, 32), (unsigned)cdiv(K, 32), (unsigned)E_);
        cvt_transpose_bf16<<<g, dim3(32, 8), 0, stream>>>(src, dst, K, N);
    };

    // 1) convert weights to bf16, transposed to N x K (once; removes GEMM-side scatters)
    TRT(Wg1, Wg1t, Dd, HG, 1);
    TRT(Wg2, Wg2t, HG, Ee, 1);
    TRT(We1, We1t, DC, HE, Ee);
    TRT(We2, We2t, HE, Zz, Ee);
    TRT(WA,  WAt,  Zz, NH, 1);
    TRT(WB,  WBt,  Zz, NH, 1);
    concat_bf16<<<(unsigned)cdiv((long long)Bb * DC, T), T, 0, stream>>>(s, c, xb, (long long)Bb * DC);

    // 2) gating MLP: g1 = relu(s@Wg1+bg1); logits = g1@Wg2+bg2; softmax
    gemm_bf16_wmma<0, false><<<dim3(HG / 128, Bb / 128), 256, 0, stream>>>(
        xb, DC, Wg1t, bg1, nullptr, g1, HG, nullptr, 0, Bb, HG, Dd);
    gemm_bf16_wmma<2, true><<<dim3(1, Bb / 128), 256, 0, stream>>>(
        g1, HG, Wg2t, bg2, logit, nullptr, Ee, nullptr, 0, Bb, Ee, HG);
    softmax16<<<(unsigned)cdiv(Bb, T), T, 0, stream>>>(logit, gates, Bb);

    // 3) soft-MoE experts, gate-weighted accumulation fused into GEMM2 epilogue
    zero_f32<<<(unsigned)cdiv((long long)Bb * Zz, T), T, 0, stream>>>(z, (long long)Bb * Zz);
    for (int e = 0; e < Ee; e++) {
        gemm_bf16_wmma<0, false><<<dim3(HE / 128, Bb / 128), 256, 0, stream>>>(
            xb, DC, We1t + (size_t)e * DC * HE, be1 + (size_t)e * HE,
            nullptr, h, HE, nullptr, 0, Bb, HE, DC);
        gemm_bf16_wmma<1, false><<<dim3(Zz / 128, Bb / 128), 256, 0, stream>>>(
            h, HE, We2t + (size_t)e * HE * Zz, be2 + (size_t)e * Zz,
            z, nullptr, Zz, gates + e, Ee, Bb, Zz, HE);
    }

    // 4) LoRA decode heads straight into d_out (A then Bm; row-major == reshape)
    cvt_f32_bf16<<<(unsigned)cdiv((long long)Bb * Zz, T), T, 0, stream>>>(z, zb, (long long)Bb * Zz);
    gemm_bf16_wmma<2, false><<<dim3(NH / 128, Bb / 128), 256, 0, stream>>>(
        zb, Zz, WAt, bA, out, nullptr, NH, nullptr, 0, Bb, NH, Zz);
    gemm_bf16_wmma<2, false><<<dim3(NH / 128, Bb / 128), 256, 0, stream>>>(
        zb, Zz, WBt, bB, out + (size_t)Bb * NH, nullptr, NH, nullptr, 0, Bb, NH, Zz);
}
